// MultiHeadAttention_50972671869112
// MI455X (gfx1250) — matrix operand order repaired and checked by reading, not compiled, not run on hardware
//
#include <hip/hip_runtime.h>

typedef __attribute__((ext_vector_type(8)))  float        v8f;
typedef __attribute__((ext_vector_type(16))) __bf16       v16bf;
typedef __attribute__((ext_vector_type(8)))  __bf16       v8bf;
typedef __attribute__((ext_vector_type(4)))  unsigned int u32x4;
typedef __attribute__((ext_vector_type(8)))  unsigned int u32x8;

#define S_LEN  1024
#define DMODEL 1024
#define NHEAD  16
#define HDIM   64

static __device__ __forceinline__ v16bf cat8(v8bf lo, v8bf hi) {
  return __builtin_shufflevector(lo, hi, 0,1,2,3,4,5,6,7,8,9,10,11,12,13,14,15);
}

static __device__ __forceinline__ v8f wmma_bf16(v16bf a, v16bf b, v8f c) {
  // 8 args: (neg_a, A, neg_b, B, c_mod, C, reuse_a, reuse_b)
  return __builtin_amdgcn_wmma_f32_16x16x32_bf16(false, a, false, b, (short)0, c,
                                                 false, false);
}

// ---------------------------------------------------------------------------
// Tensor Data Mover: 2D tile (bf16 elements) global -> LDS.
// D# group0: count=1 | lds_addr | global_addr[56:0] | type=2
// D# group1: data_size=2B, tensor_dim0/1 = tile_dim0/1 (tight), dim0 stride.
// 2-SGPR-group form: VADDR2/VADDR3 = NULL (tensors up to 2D).
// ---------------------------------------------------------------------------
static __device__ __forceinline__ void tdm_load_2d_bf16(unsigned lds_off,
                                                        const void* gsrc,
                                                        unsigned tile_d0,   // elems/row
                                                        unsigned tile_d1,   // rows
                                                        unsigned long long row_stride) // elems
{
  unsigned long long ga = (unsigned long long)gsrc;
  u32x4 g0;
  g0.x = 1u;                                                 // count=1, user mode
  g0.y = lds_off;                                            // lds_addr (bytes)
  g0.z = (unsigned)ga;                                       // global_addr[31:0]
  g0.w = (unsigned)((ga >> 32) & 0x01FFFFFFu) | (2u << 30);  // addr[56:32], type=2
  u32x8 g1;
  g1[0] = (1u << 16);                                        // data_size=1 -> 2 bytes
  g1[1] = (tile_d0 & 0xFFFFu) << 16;                         // tensor_dim0[15:0]
  g1[2] = (tile_d0 >> 16) | ((tile_d1 & 0xFFFFu) << 16);     // dim0 hi | tensor_dim1 lo
  g1[3] = (tile_d1 >> 16) | ((tile_d0 & 0xFFFFu) << 16);     // dim1 hi | tile_dim0
  g1[4] = tile_d1 & 0xFFFFu;                                 // tile_dim1 (tile_dim2=0)
  g1[5] = (unsigned)row_stride;                              // tensor_dim0_stride lo
  g1[6] = (unsigned)(row_stride >> 32) & 0xFFFFu;            // stride hi
  g1[7] = 0u;
  asm volatile("tensor_load_to_lds %0, %1" :: "s"(g0), "s"(g1) : "memory");
}

static __device__ __forceinline__ unsigned lds_offset_of(const void* p) {
  return (unsigned)(unsigned long long)p;  // LDS aperture lives in addr[63:32]
}

// ---------------------------------------------------------------------------
// Tiled GEMM: C[64x64] per block, 4 waves, K-step 32, bf16 WMMA, fp32 accum.
// modes: 0=Q proj (bias, *0.125, -> qws[B,H,S,DK] bf16)
//        1=K proj (bias + pos_k,  -> kws[B,H,S,DK] bf16)
//        2=V proj (bias + pos_v,  -> vtws[B,H,DV,S] bf16, transposed)
//        3=O proj (bias + residual Q, -> out_f[B*S,D] fp32)
// ---------------------------------------------------------------------------
__global__ __launch_bounds__(128)
void gemm64_kernel(const void* __restrict__ Aptr, int a_bf16,
                   const float* __restrict__ W,
                   const float* __restrict__ bias,
                   const float* __restrict__ pos,
                   const float* __restrict__ resid,
                   __bf16* __restrict__ out_bf,
                   float* __restrict__ out_f,
                   int M, int N, int K, int mode)
{
  constexpr int LDA = 48; // padded stride (96B: multiple of 32B -> aligned frags)
  __shared__ __align__(32) __bf16 As[64 * LDA];   // A tile, row-major
  __shared__ __align__(32) __bf16 Bs[64 * LDA];   // W tile, col-major [n][k]

  const int tiles_n = N >> 6;
  const int bm = blockIdx.x / tiles_n;
  const int bn = blockIdx.x % tiles_n;
  const int t    = threadIdx.x;
  const int w    = t >> 5;
  const int lane = t & 31;
  const int ln   = lane & 15;
  const int kh   = lane >> 4;

  const int arow  = t >> 1;          // 0..63
  const int ahalf = (t & 1) * 16;    // 0 or 16
  const int wk    = t >> 2;          // 0..31
  const int wn0   = (t & 3) * 16;    // 0,16,32,48

  v8f acc[4] = {};

  for (int kb = 0; kb < K; kb += 32) {
    __syncthreads();
    // ---- stage A tile (convert fp32->bf16 unless already bf16) ----
    if (a_bf16) {
      const v8bf* ap = (const v8bf*)((const __bf16*)Aptr +
                          (size_t)(bm * 64 + arow) * K + kb + ahalf);
      v8bf x0 = ap[0], x1 = ap[1];
      *(v8bf*)&As[arow * LDA + ahalf]     = x0;
      *(v8bf*)&As[arow * LDA + ahalf + 8] = x1;
    } else {
      const float4* ap = (const float4*)((const float*)Aptr +
                          (size_t)(bm * 64 + arow) * K + kb + ahalf);
#pragma unroll
      for (int i = 0; i < 4; ++i) {
        float4 v = ap[i];
        __bf16* d = &As[arow * LDA + ahalf + 4 * i];
        d[0] = (__bf16)v.x; d[1] = (__bf16)v.y;
        d[2] = (__bf16)v.z; d[3] = (__bf16)v.w;
      }
    }
    // ---- stage W tile transposed into Bs[n][k] ----
    {
      const float4* wp = (const float4*)(W + (size_t)(kb + wk) * N + bn * 64 + wn0);
#pragma unroll
      for (int i = 0; i < 4; ++i) {
        float4 v = wp[i];
        Bs[(wn0 + 4 * i + 0) * LDA + wk] = (__bf16)v.x;
        Bs[(wn0 + 4 * i + 1) * LDA + wk] = (__bf16)v.y;
        Bs[(wn0 + 4 * i + 2) * LDA + wk] = (__bf16)v.z;
        Bs[(wn0 + 4 * i + 3) * LDA + wk] = (__bf16)v.w;
      }
    }
    // prefetch next K tile into cache while we compute this one
    if (kb + 32 < K) {
      if (a_bf16)
        __builtin_prefetch((const __bf16*)Aptr +
            (size_t)(bm * 64 + arow) * K + kb + 32 + ahalf, 0, 1);
      else
        __builtin_prefetch((const float*)Aptr +
            (size_t)(bm * 64 + arow) * K + kb + 32 + ahalf, 0, 1);
      __builtin_prefetch(W + (size_t)(kb + 32 + wk) * N + bn * 64 + wn0, 0, 1);
    }
    __syncthreads();

    // ---- A fragment (16x32): lane row m=ln, K run [kh*16,+16)
    v8bf alo = *(const v8bf*)&As[(16 * w + ln) * LDA + kh * 16];
    v8bf ahi = *(const v8bf*)&As[(16 * w + ln) * LDA + kh * 16 + 8];
    v16bf a  = cat8(alo, ahi);
    // ---- B fragments into independent registers, then WMMA back-to-back ----
    v16bf b0 = *(const v16bf*)&Bs[(ln)      * LDA + kh * 16];
    v16bf b1 = *(const v16bf*)&Bs[(16 + ln) * LDA + kh * 16];
    v16bf b2 = *(const v16bf*)&Bs[(32 + ln) * LDA + kh * 16];
    v16bf b3 = *(const v16bf*)&Bs[(48 + ln) * LDA + kh * 16];
    acc[0] = wmma_bf16(a, b0, acc[0]);
    acc[1] = wmma_bf16(a, b1, acc[1]);
    acc[2] = wmma_bf16(a, b2, acc[2]);
    acc[3] = wmma_bf16(a, b3, acc[3]);
  }

  // ---- epilogue (C layout: VGPR r -> row r+8*kh, lane ln -> col) ----
#pragma unroll
  for (int c = 0; c < 4; ++c) {
#pragma unroll
    for (int r = 0; r < 8; ++r) {
      int gm = bm * 64 + 16 * w + r + 8 * kh;
      int gn = bn * 64 + 16 * c + ln;
      float val = acc[c][r] + bias[gn];
      if (mode == 0) {
        int b = gm >> 10, s = gm & 1023, h = gn >> 6, dk = gn & 63;
        out_bf[(((size_t)(b * NHEAD + h)) * S_LEN + s) * HDIM + dk] =
            (__bf16)(val * 0.125f); // fold 1/sqrt(DK) into q
      } else if (mode == 1) {
        int b = gm >> 10, s = gm & 1023, h = gn >> 6, dk = gn & 63;
        val += pos[((size_t)(h * S_LEN + s)) * HDIM + dk];
        out_bf[(((size_t)(b * NHEAD + h)) * S_LEN + s) * HDIM + dk] = (__bf16)val;
      } else if (mode == 2) {
        int b = gm >> 10, s = gm & 1023, h = gn >> 6, dv = gn & 63;
        val += pos[((size_t)(h * S_LEN + s)) * HDIM + dv];
        out_bf[(((size_t)(b * NHEAD + h)) * HDIM + dv) * S_LEN + s] = (__bf16)val;
      } else {
        val += resid[(size_t)gm * DMODEL + gn];
        out_f[(size_t)gm * DMODEL + gn] = val;
      }
    }
  }
}

// ---------------------------------------------------------------------------
// Attention: block = one (b,h) x 64 query rows; each wave owns 16 rows.
// K/V chunks (32 keys) are DMA'd into double-buffered LDS by the Tensor Data
// Mover (wave 0 issues, s_wait_tensorcnt + barrier synchronizes), overlapping
// chunk i+1 DMA with WMMA on chunk i.
// Pass 1: exact online softmax stats. Pass 2: recompute scores, write the
// normalized probs (the 256MB output that bounds this workload), P@V via WMMA.
// ---------------------------------------------------------------------------
__global__ __launch_bounds__(128)
void attn_kernel(const __bf16* __restrict__ qws, const __bf16* __restrict__ kws,
                 const __bf16* __restrict__ vtws, __bf16* __restrict__ ctxws,
                 float* __restrict__ attn_out)
{
  __shared__ __align__(32) __bf16 Ks[2][32 * HDIM];  // [key][dk]   4KB x2
  __shared__ __align__(32) __bf16 Vs[2][HDIM * 32];  // [dv][key]   4KB x2
  __shared__ __align__(32) __bf16 pls[4][16 * 32];   // per-wave P tile

  const int bh   = blockIdx.x >> 4;  // 0..63 = b*16+h
  const int qt   = blockIdx.x & 15;
  const int t    = threadIdx.x;
  const int w    = t >> 5;
  const int lane = t & 31;
  const int ln   = lane & 15;
  const int kh   = lane >> 4;
  const int b    = bh >> 4;
  const int h    = bh & 15;
  const int q0   = qt * 64 + w * 16;

  const __bf16* kbase = kws  + (size_t)bh * S_LEN * HDIM;
  const __bf16* vbase = vtws + (size_t)bh * HDIM * S_LEN;
  const unsigned ks_off[2] = { lds_offset_of(&Ks[0][0]), lds_offset_of(&Ks[1][0]) };
  const unsigned vs_off[2] = { lds_offset_of(&Vs[0][0]), lds_offset_of(&Vs[1][0]) };

  // Q A-fragments for dk halves 0..31 / 32..63 (kept resident)
  const __bf16* qrowp = qws + ((size_t)bh * S_LEN + q0 + ln) * HDIM;
  v16bf qa[2];
#pragma unroll
  for (int s = 0; s < 2; ++s) {
    v8bf lo = *(const v8bf*)(qrowp + 32 * s + kh * 16);
    v8bf hi = *(const v8bf*)(qrowp + 32 * s + kh * 16 + 8);
    qa[s] = cat8(lo, hi);
  }

  float mrow[8], lrow[8];
#pragma unroll
  for (int r = 0; r < 8; ++r) { mrow[r] = -1e30f; lrow[r] = 0.0f; }

  // ---------------- Pass 1: softmax statistics ----------------
  if (w == 0)
    tdm_load_2d_bf16(ks_off[0], kbase, HDIM, 32, HDIM);      // chunk 0 -> buf 0
  for (int ch = 0; ch < 32; ++ch) {
    const int buf = ch & 1;
    if (w == 0) __builtin_amdgcn_s_wait_tensorcnt(0);
    __syncthreads();                                          // buf ready, prev reads done
    if (w == 0 && ch + 1 < 32)
      tdm_load_2d_bf16(ks_off[1 - buf],
                       kbase + (size_t)(ch + 1) * 32 * HDIM, HDIM, 32, HDIM);

    v16bf k00 = *(const v16bf*)&Ks[buf][(ln)      * HDIM + kh * 16];
    v16bf k01 = *(const v16bf*)&Ks[buf][(ln)      * HDIM + 32 + kh * 16];
    v16bf k10 = *(const v16bf*)&Ks[buf][(16 + ln) * HDIM + kh * 16];
    v16bf k11 = *(const v16bf*)&Ks[buf][(16 + ln) * HDIM + 32 + kh * 16];
    v8f sa[2] = {};
    sa[0] = wmma_bf16(qa[0], k00, sa[0]);
    sa[0] = wmma_bf16(qa[1], k01, sa[0]);
    sa[1] = wmma_bf16(qa[0], k10, sa[1]);
    sa[1] = wmma_bf16(qa[1], k11, sa[1]);

#pragma unroll
    for (int r = 0; r < 8; ++r) {
      float tmx = fmaxf(sa[0][r], sa[1][r]);
      tmx = fmaxf(tmx, __shfl_xor(tmx, 1));
      tmx = fmaxf(tmx, __shfl_xor(tmx, 2));
      tmx = fmaxf(tmx, __shfl_xor(tmx, 4));
      tmx = fmaxf(tmx, __shfl_xor(tmx, 8));
      float mn = fmaxf(mrow[r], tmx);
      float es = __expf(sa[0][r] - mn) + __expf(sa[1][r] - mn);
      es += __shfl_xor(es, 1);
      es += __shfl_xor(es, 2);
      es += __shfl_xor(es, 4);
      es += __shfl_xor(es, 8);
      lrow[r] = lrow[r] * __expf(mrow[r] - mn) + es;
      mrow[r] = mn;
    }
    __syncthreads();                                          // done reading buf
  }
  float linv[8];
#pragma unroll
  for (int r = 0; r < 8; ++r) linv[r] = 1.0f / lrow[r];

  // ---------------- Pass 2: probs out + context ----------------
  v8f cacc[4] = {};
  if (w == 0) {
    tdm_load_2d_bf16(ks_off[0], kbase, HDIM, 32, HDIM);
    tdm_load_2d_bf16(vs_off[0], vbase, 32, HDIM, S_LEN);
  }
  for (int ch = 0; ch < 32; ++ch) {
    const int buf = ch & 1;
    const int kc  = ch * 32;
    if (w == 0) __builtin_amdgcn_s_wait_tensorcnt(0);
    __syncthreads();
    if (w == 0 && ch + 1 < 32) {
      tdm_load_2d_bf16(ks_off[1 - buf],
                       kbase + (size_t)(ch + 1) * 32 * HDIM, HDIM, 32, HDIM);
      tdm_load_2d_bf16(vs_off[1 - buf],
                       vbase + (size_t)(ch + 1) * 32, 32, HDIM, S_LEN);
    }

    v16bf k00 = *(const v16bf*)&Ks[buf][(ln)      * HDIM + kh * 16];
    v16bf k01 = *(const v16bf*)&Ks[buf][(ln)      * HDIM + 32 + kh * 16];
    v16bf k10 = *(const v16bf*)&Ks[buf][(16 + ln) * HDIM + kh * 16];
    v16bf k11 = *(const v16bf*)&Ks[buf][(16 + ln) * HDIM + 32 + kh * 16];
    v8f sa[2] = {};
    sa[0] = wmma_bf16(qa[0], k00, sa[0]);
    sa[0] = wmma_bf16(qa[1], k01, sa[0]);
    sa[1] = wmma_bf16(qa[0], k10, sa[1]);
    sa[1] = wmma_bf16(qa[1], k11, sa[1]);

#pragma unroll
    for (int c = 0; c < 2; ++c)
#pragma unroll
      for (int r = 0; r < 8; ++r) {
        float p = __expf(sa[c][r] - mrow[r]) * linv[r];
        int qr = q0 + r + 8 * kh;
        attn_out[((size_t)bh * S_LEN + qr) * S_LEN + kc + 16 * c + ln] = p;
        pls[w][(r + 8 * kh) * 32 + 16 * c + ln] = (__bf16)p;
      }
    __syncthreads();
    // P as A fragment (16 rows x 32 keys)
    v8bf plo = *(const v8bf*)&pls[w][ln * 32 + kh * 16];
    v8bf phi = *(const v8bf*)&pls[w][ln * 32 + kh * 16 + 8];
    v16bf pa = cat8(plo, phi);
    // V^T frags: lane col = dv, key run [kh*16,+16) contiguous in LDS
    v16bf vb0 = *(const v16bf*)&Vs[buf][(ln)      * 32 + kh * 16];
    v16bf vb1 = *(const v16bf*)&Vs[buf][(16 + ln) * 32 + kh * 16];
    v16bf vb2 = *(const v16bf*)&Vs[buf][(32 + ln) * 32 + kh * 16];
    v16bf vb3 = *(const v16bf*)&Vs[buf][(48 + ln) * 32 + kh * 16];
    cacc[0] = wmma_bf16(pa, vb0, cacc[0]);
    cacc[1] = wmma_bf16(pa, vb1, cacc[1]);
    cacc[2] = wmma_bf16(pa, vb2, cacc[2]);
    cacc[3] = wmma_bf16(pa, vb3, cacc[3]);
  }

  // context -> ctxws[b*S+qr][h*64+dv] bf16 (A operand of output projection)
#pragma unroll
  for (int c = 0; c < 4; ++c)
#pragma unroll
    for (int r = 0; r < 8; ++r) {
      int qr = q0 + r + 8 * kh;
      int dv = 16 * c + ln;
      ctxws[((size_t)b * S_LEN + qr) * DMODEL + h * HDIM + dv] = (__bf16)cacc[c][r];
    }
}

// ---------------------------------------------------------------------------
// Row layernorm over D=1024: one block per row.
// ---------------------------------------------------------------------------
__global__ __launch_bounds__(256)
void ln_kernel(const float* __restrict__ x, const float* __restrict__ gamma,
               const float* __restrict__ beta, float* __restrict__ out)
{
  __shared__ float red[256];
  __shared__ float red2[256];
  const int row = blockIdx.x;
  const int t   = threadIdx.x;
  const float4* xp = (const float4*)(x + (size_t)row * DMODEL);
  float4 v = xp[t];
  red[t]  = v.x + v.y + v.z + v.w;
  red2[t] = v.x * v.x + v.y * v.y + v.z * v.z + v.w * v.w;
  __syncthreads();
  for (int off = 128; off > 0; off >>= 1) {
    if (t < off) { red[t] += red[t + off]; red2[t] += red2[t + off]; }
    __syncthreads();
  }
  float mu  = red[0] * (1.0f / 1024.0f);
  float var = red2[0] * (1.0f / 1024.0f) - mu * mu;
  float rs  = rsqrtf(var + 1e-5f);
  float4 g  = ((const float4*)gamma)[t];
  float4 be = ((const float4*)beta)[t];
  float4 o;
  o.x = (v.x - mu) * rs * g.x + be.x;
  o.y = (v.y - mu) * rs * g.y + be.y;
  o.z = (v.z - mu) * rs * g.z + be.z;
  o.w = (v.w - mu) * rs * g.w + be.w;
  ((float4*)(out + (size_t)row * DMODEL))[t] = o;
}

extern "C" void kernel_launch(void* const* d_in, const int* in_sizes, int n_in,
                              void* d_out, int out_size, void* d_ws, size_t ws_size,
                              hipStream_t stream) {
  const float* Q     = (const float*)d_in[0];
  const float* Kin   = (const float*)d_in[1];
  const float* V     = (const float*)d_in[2];
  const float* Wq    = (const float*)d_in[3];
  const float* bq    = (const float*)d_in[4];
  const float* Wk    = (const float*)d_in[5];
  const float* bk    = (const float*)d_in[6];
  const float* Wv    = (const float*)d_in[7];
  const float* bv    = (const float*)d_in[8];
  const float* Wo    = (const float*)d_in[9];
  const float* bo    = (const float*)d_in[10];
  const float* pos_k = (const float*)d_in[11];
  const float* pos_v = (const float*)d_in[12];
  const float* gamma = (const float*)d_in[13];
  const float* beta  = (const float*)d_in[14];

  float* outp  = (float*)d_out;                       // (B,S,D) fp32
  float* attnp = outp + (size_t)4 * 1024 * 1024;      // (B,H,S,S) fp32

  char* ws = (char*)d_ws;
  __bf16* qws   = (__bf16*)(ws);                      //  8 MB [B,H,S,DK]
  __bf16* kws   = (__bf16*)(ws + ((size_t)8  << 20)); //  8 MB [B,H,S,DK]
  __bf16* vtws  = (__bf16*)(ws + ((size_t)16 << 20)); //  8 MB [B,H,DV,S]
  __bf16* ctxws = (__bf16*)(ws + ((size_t)24 << 20)); //  8 MB [B*S,H*DV]
  float*  tmp   = (float*)(ws + ((size_t)32 << 20));  // 16 MB pre-LN fp32

  const int M = 4096, N = 1024, Kd = 1024;
  dim3 blk(128);
  dim3 grd((M / 64) * (N / 64));
  gemm64_kernel<<<grd, blk, 0, stream>>>(Q,   0, Wq, bq, nullptr, nullptr,
                                         qws,  nullptr, M, N, Kd, 0);
  gemm64_kernel<<<grd, blk, 0, stream>>>(Kin, 0, Wk, bk, pos_k,  nullptr,
                                         kws,  nullptr, M, N, Kd, 1);
  gemm64_kernel<<<grd, blk, 0, stream>>>(V,   0, Wv, bv, pos_v,  nullptr,
                                         vtws, nullptr, M, N, Kd, 2);
  attn_kernel<<<dim3(64 * 16), blk, 0, stream>>>(qws, kws, vtws, ctxws, attnp);
  gemm64_kernel<<<grd, blk, 0, stream>>>(ctxws, 1, Wo, bo, nullptr, Q,
                                         nullptr, tmp, M, N, Kd, 3);
  ln_kernel<<<dim3(4096), dim3(256), 0, stream>>>(tmp, gamma, beta, outp);
}
